// LumiFeat_61813169324870
// MI455X (gfx1250) — matrix operand order repaired and checked by reading, not compiled, not run on hardware
//
#include <hip/hip_runtime.h>
#include <hip/hip_bf16.h>
#include <math.h>

typedef __attribute__((ext_vector_type(16))) _Float16 v16h;
typedef __attribute__((ext_vector_type(8)))  _Float16 v8h;
typedef __attribute__((ext_vector_type(8)))  float    v8f;

#define HW    512
#define NPIX  (HW * HW)        // 262144 = 2^18
#define NIMG  4
#define TOTAL (NIMG * NPIX)    // 1048576

union U16H { v16h v; v8h p[2]; _Float16 h[16]; };
union U8F  { v8f  v; float    f[8];  };

__device__ inline void atomicMinF(float* addr, float val) {
    unsigned* a = (unsigned*)addr;
    unsigned old = __float_as_uint(*addr);
    while (val < __uint_as_float(old)) {
        unsigned assumed = old;
        old = atomicCAS(a, assumed, __float_as_uint(val));
        if (old == assumed) break;
    }
}
__device__ inline void atomicMaxF(float* addr, float val) {
    unsigned* a = (unsigned*)addr;
    unsigned old = __float_as_uint(*addr);
    while (val > __uint_as_float(old)) {
        unsigned assumed = old;
        old = atomicCAS(a, assumed, __float_as_uint(val));
        if (old == assumed) break;
    }
}

// exact count of zero bytes in a 64-bit word (no cross-byte borrow artifacts)
__device__ inline int zbytes(unsigned long long x) {
    unsigned long long t = ((x & 0x7F7F7F7F7F7F7F7Full) + 0x7F7F7F7F7F7F7F7Full) | x;
    return __popcll(~t & 0x8080808080808080ull);
}

// ---------------- init workspace ----------------
__global__ void __launch_bounds__(256) lumifeat_init(unsigned* hist, float* smin, float* smax) {
    int t = threadIdx.x;
    hist[t] = 0u;                       // 256 = 4 images * 64 bins
    if (t < NIMG) { smin[t] = 3.4e38f; smax[t] = -3.4e38f; }
}

// ---------------- per-image global luminance histogram ----------------
__global__ void __launch_bounds__(256) lumifeat_hist(const float* __restrict__ in,
                                                     unsigned* __restrict__ hist) {
    __shared__ unsigned lh[64];
    const int tid = threadIdx.x;
    const int img = blockIdx.y;
    if (tid < 64) lh[tid] = 0u;
    __syncthreads();
    const float* base = in + (size_t)img * 3 * NPIX;   // L channel
    for (int i = blockIdx.x * blockDim.x + tid; i < NPIX; i += gridDim.x * blockDim.x) {
        float Ln = fminf(fmaxf(base[i] * 0.01f, 0.0f), 1.0f);
        int bi = (int)(Ln * 64.0f);
        bi = bi < 0 ? 0 : (bi > 63 ? 63 : bi);
        atomicAdd(&lh[bi], 1u);
    }
    __syncthreads();
    if (tid < 64) atomicAdd(&hist[img * 64 + tid], lh[tid]);
}

// ---------------- fused main kernel: 16x16 tile per block ----------------
__global__ void __launch_bounds__(256) lumifeat_main(const float* __restrict__ in,
                                                     const unsigned* __restrict__ hist,
                                                     float* __restrict__ out,
                                                     float* __restrict__ smin,
                                                     float* __restrict__ smax) {
    __shared__ __align__(16) unsigned char s_bins[32 * 32];   // row-major [hy][hx]
    __shared__ __align__(32) _Float16 s_acm[32 * 32];         // column-major [hx][hy], zero padded
    __shared__ __align__(32) _Float16 s_bcm[32 * 32];         // column-major [hx][hy], zero padded
    __shared__ __align__(32) _Float16 s_v[16 * 32];           // vertical box sums, row-major [row][col]
    __shared__ float s_mu[2][16 * 16];                        // raw 15x15 box sums of a,b
    __shared__ float s_red[2][8];

    const int tid = threadIdx.x;
    const int tx0 = blockIdx.x * 16;
    const int ty0 = blockIdx.y * 16;
    const int img = blockIdx.z;
    const float* base = in + (size_t)img * 3 * NPIX;

    // ---- load 30x30 halo into 32x32 padded tiles (exactly 4 iterations, no division) ----
    #pragma unroll
    for (int it = 0; it < 4; ++it) {
        int i = tid + it * 256;
        int hx = i & 31, hy = i >> 5;
        int gy = ty0 + hy - 7;
        int gx = tx0 + hx - 7;
        unsigned char bb = 255;                 // sentinel: never matches a bin (bins are 0..63)
        _Float16 av = (_Float16)0.0f, bv = (_Float16)0.0f;   // zero padding == reference zero-pad
        if (hy < 30 && hx < 30 && gy >= 0 && gy < HW && gx >= 0 && gx < HW) {
            float Lh = fminf(fmaxf(base[gy * HW + gx] * 0.01f, 0.0f), 1.0f);
            int bi = (int)(Lh * 64.0f);
            bi = bi < 0 ? 0 : (bi > 63 ? 63 : bi);
            bb = (unsigned char)bi;
            av = (_Float16)((base[NPIX + gy * HW + gx] + 128.0f) * (1.0f / 255.0f));
            bv = (_Float16)((base[2 * NPIX + gy * HW + gx] + 128.0f) * (1.0f / 255.0f));
        }
        s_bins[hy * 32 + hx] = bb;
        s_acm[hx * 32 + hy] = av;
        s_bcm[hx * 32 + hy] = bv;
    }
    __syncthreads();

    // ---- wave 0: separable 15-tap box sums as banded GEMMs via WMMA ----
    if (tid < 32) {
        const int lane = tid;
        const int half = lane >> 4;
        const int m = lane & 15;           // A row / B,C,D column
        U16H amv, bmh;
        #pragma unroll
        for (int e = 0; e < 16; ++e) {
            // A-layout K index (contiguous, matches column-major B load)
            int Ka = 16 * half + e;
            amv.h[e] = (_Float16)((Ka >= m && Ka <= m + 14) ? 1.0f : 0.0f);   // Mv[i][k]
            // B-layout K index (table order, matches afrag)
            int Kb = (e < 8) ? (8 * half + e) : (16 + 8 * half + (e - 8));
            bmh.h[e] = (_Float16)((Kb >= m && Kb <= m + 14) ? 1.0f : 0.0f);   // Mh[k][n]
        }
        #pragma unroll
        for (int ch = 0; ch < 2; ++ch) {
            const _Float16* tile = ch ? s_bcm : s_acm;
            // vertical: V(16x32) = Mv(16x32) x Tile(32x32), two 16-col chunks
            #pragma unroll
            for (int c = 0; c < 2; ++c) {
                const int col = 16 * c + m;
                U16H bfrag;
                // 16 contiguous halves per lane: column-major tile, 32B aligned
                bfrag.v = *(const v16h*)(tile + col * 32 + 16 * half);
                U8F acc{};
                acc.v = __builtin_amdgcn_wmma_f32_16x16x32_f16(
                    false, amv.v, false, bfrag.v, (short)0, acc.v, false, false);
                #pragma unroll
                for (int r = 0; r < 8; ++r)
                    s_v[(r + 8 * half) * 32 + col] = (_Float16)acc.f[r];
            }
            // horizontal: Box(16x16) = V(16x32) x Mh(32x16)
            U16H afrag;
            afrag.p[0] = *(const v8h*)(s_v + m * 32 + 8 * half);        // K = 8*half + 0..7
            afrag.p[1] = *(const v8h*)(s_v + m * 32 + 16 + 8 * half);   // K = 16 + 8*half + 0..7
            U8F acc2{};
            acc2.v = __builtin_amdgcn_wmma_f32_16x16x32_f16(
                false, afrag.v, false, bmh.v, (short)0, acc2.v, false, false);
            #pragma unroll
            for (int r = 0; r < 8; ++r)
                s_mu[ch][(r + 8 * half) * 16 + m] = acc2.f[r];
        }
    }
    __syncthreads();

    // ---- per-pixel: own-bin window count (packed u64 compares), surprise, L_perp ----
    const int txl = tid & 15, tyl = tid >> 4;
    const int x = tx0 + txl, y = ty0 + tyl;

    float Ln = fminf(fmaxf(base[y * HW + x] * 0.01f, 0.0f), 1.0f);
    float an = (base[NPIX + y * HW + x] + 128.0f) * (1.0f / 255.0f);
    float bn = (base[2 * NPIX + y * HW + x] + 128.0f) * (1.0f / 255.0f);
    int bin = (int)(Ln * 64.0f);
    bin = bin < 0 ? 0 : (bin > 63 ? 63 : bin);

    const int q0 = txl >> 3;                 // starting qword in the 32B row
    const int sh = (txl & 7) * 8;            // bit shift within qword
    const unsigned long long bs = 0x0101010101010101ull * (unsigned long long)(unsigned)bin;
    int cnt = 0;
    #pragma unroll 5
    for (int dy = 0; dy < 15; ++dy) {
        const unsigned long long* row =
            (const unsigned long long*)(s_bins + (tyl + dy) * 32) + q0;
        unsigned long long a0 = row[0], a1 = row[1], a2 = row[2];
        unsigned long long w0 = sh ? ((a0 >> sh) | (a1 << (64 - sh))) : a0;  // bytes dx 0..7
        unsigned long long w1 = sh ? ((a1 >> sh) | (a2 << (64 - sh))) : a1;  // bytes dx 8..15
        unsigned long long x0 = w0 ^ bs;
        unsigned long long x1 = (w1 ^ bs) | 0xFF00000000000000ull;           // drop dx==15
        cnt += zbytes(x0) + zbytes(x1);
    }

    int nH = min(x + 7, HW - 1) - max(x - 7, 0) + 1;
    int nV = min(y + 7, HW - 1) - max(y - 7, 0) + 1;
    float n = (float)(nH * nV);
    float p_loc = (float)cnt / n;
    float p_glob = (float)hist[img * 64 + bin] * (1.0f / (float)NPIX);
    float S = -logf(0.9f * p_loc + 0.1f * p_glob + 1e-6f);

    float mu_a = s_mu[0][tyl * 16 + txl] / n;
    float mu_b = s_mu[1][tyl * 16 + txl] / n;
    float Lp = Ln - 0.5f * (an - mu_a) - 0.5f * (bn - mu_b);
    Lp = fminf(fmaxf(Lp, 0.0f), 1.0f);

    out[(size_t)img * NPIX + y * HW + x] = Lp;
    out[(size_t)TOTAL + (size_t)img * NPIX + y * HW + x] = S;   // raw S, normalized later

    // ---- wave32 shuffle reduce, then 8 partials -> global atomics ----
    float vmn = S, vmx = S;
    #pragma unroll
    for (int off = 16; off > 0; off >>= 1) {
        vmn = fminf(vmn, __shfl_xor(vmn, off, 32));
        vmx = fmaxf(vmx, __shfl_xor(vmx, off, 32));
    }
    if ((tid & 31) == 0) {
        s_red[0][tid >> 5] = vmn;
        s_red[1][tid >> 5] = vmx;
    }
    __syncthreads();
    if (tid == 0) {
        float mn = s_red[0][0], mx = s_red[1][0];
        #pragma unroll
        for (int w = 1; w < 8; ++w) {
            mn = fminf(mn, s_red[0][w]);
            mx = fmaxf(mx, s_red[1][w]);
        }
        atomicMinF(&smin[img], mn);
        atomicMaxF(&smax[img], mx);
    }
}

// ---------------- normalize S in place ----------------
__global__ void __launch_bounds__(256) lumifeat_norm(float* __restrict__ S,
                                                     const float* __restrict__ smin,
                                                     const float* __restrict__ smax) {
    int i = blockIdx.x * blockDim.x + threadIdx.x;
    if (i < TOTAL) {
        int img = i >> 18;   // NPIX = 2^18
        float mn = smin[img], mx = smax[img];
        S[i] = (S[i] - mn) / (mx - mn + 1e-6f);
    }
}

extern "C" void kernel_launch(void* const* d_in, const int* in_sizes, int n_in,
                              void* d_out, int out_size, void* d_ws, size_t ws_size,
                              hipStream_t stream) {
    (void)in_sizes; (void)n_in; (void)out_size; (void)ws_size;
    const float* in = (const float*)d_in[0];
    float* out = (float*)d_out;
    unsigned* hist = (unsigned*)d_ws;          // 4*64 u32
    float* smin = (float*)d_ws + 256;          // 4 floats
    float* smax = (float*)d_ws + 260;          // 4 floats

    lumifeat_init<<<1, 256, 0, stream>>>(hist, smin, smax);
    lumifeat_hist<<<dim3(64, NIMG), 256, 0, stream>>>(in, hist);
    lumifeat_main<<<dim3(HW / 16, HW / 16, NIMG), 256, 0, stream>>>(in, hist, out, smin, smax);
    lumifeat_norm<<<(TOTAL + 255) / 256, 256, 0, stream>>>(out + TOTAL, smin, smax);
}
